// GAT_86019605004484
// MI455X (gfx1250) — matrix operand order repaired and checked by reading, not compiled, not run on hardware
//
#include <hip/hip_runtime.h>
#include <hip/hip_bf16.h>

typedef __attribute__((ext_vector_type(2)))  float        v2f;
typedef __attribute__((ext_vector_type(8)))  float        v8f;
typedef __attribute__((ext_vector_type(16))) __bf16       v16bf;
typedef __attribute__((ext_vector_type(4)))  unsigned int v4u;
typedef __attribute__((ext_vector_type(4)))  int          v4i;
typedef __attribute__((ext_vector_type(8)))  int          v8i;

#define EPSLN  1e-5f
#define NSLOPE 0.2f
#define BB 8
#define NN 1024
#define DD 64
#define HH 4

#if __has_builtin(__builtin_amdgcn_tensor_load_to_lds)
#define HAVE_TDM 1
#endif

__device__ __forceinline__ unsigned short f2bf(float x){
  unsigned int u = __float_as_uint(x);
  u += 0x7FFFu + ((u >> 16) & 1u);           // round-to-nearest-even
  return (unsigned short)(u >> 16);
}
__device__ __forceinline__ unsigned rl(unsigned x){
  return (unsigned)__builtin_amdgcn_readfirstlane((int)x);
}

union BF16x16 { unsigned short u[16]; uint4 q[2]; v16bf v; };

#if defined(HAVE_TDM)
// Issue a TDM 2D tile load: FH rows x 32 bf16, row stride NN*2B -> LDS.
template<int FH>
__device__ __forceinline__ void tdm_issue(const unsigned short* gp, unsigned ldsOff){
  unsigned long long ga = (unsigned long long)(uintptr_t)gp;
  unsigned g0w0 = rl(1u);                                  // count=1
  unsigned g0w1 = rl(ldsOff);                              // lds_addr
  unsigned g0w2 = rl((unsigned)ga);                        // global_addr lo
  unsigned g0w3 = rl(((unsigned)(ga >> 32) & 0x01FFFFFFu) | 0x80000000u); // hi | type=2
  v4u g0 = { g0w0, g0w1, g0w2, g0w3 };
  unsigned g1w0 = rl(1u << 16);                 // data_size=2B, workgroup_mask=0
  unsigned g1w1 = rl((unsigned)NN << 16);       // tensor_dim0 = N (lo16)
  unsigned g1w2 = rl((unsigned)FH << 16);       // tensor_dim1 = FH (lo16)
  unsigned g1w3 = rl(32u << 16);                // tile_dim0 = 32
  unsigned g1w4 = rl((unsigned)FH);             // tile_dim1 = FH, tile_dim2 = 0
  unsigned g1w5 = rl((unsigned)NN);             // tensor_dim0_stride = N
  v8i g1 = { (int)g1w0, (int)g1w1, (int)g1w2, (int)g1w3,
             (int)g1w4, (int)g1w5, 0, 0 };
  v4i gz = { 0, 0, 0, 0 };
#if defined(__clang_major__) && (__clang_major__ >= 23)
  v8i gz8 = { 0, 0, 0, 0, 0, 0, 0, 0 };
  __builtin_amdgcn_tensor_load_to_lds(g0, g1, gz, gz, gz8, 0);
#else
  __builtin_amdgcn_tensor_load_to_lds(g0, g1, gz, gz, 0);
#endif
}
#endif

// ---------------------------------------------------------------------------
// Dense fp32 GEMM via V_WMMA_F32_16X16X4_F32. One wave per 16x16 C tile.
// ---------------------------------------------------------------------------
__global__ void gemm_f32_wmma(const float* __restrict__ A, const float* __restrict__ Bm,
                              float* __restrict__ C, int K, int Nc){
  const int lane = threadIdx.x & 31;
  const int wave = threadIdx.x >> 5;
  const int tm   = blockIdx.x;
  const int tn   = blockIdx.y * 4 + wave;
  const int mrow = tm * 16 + (lane & 15);
  const int col  = tn * 16 + (lane & 15);
  const int kb   = (lane >> 4) * 2;        // lanes 16-31 hold K+2,K+3
  v8f acc = {0.f,0.f,0.f,0.f,0.f,0.f,0.f,0.f};
  for (int kk = 0; kk < K; kk += 4){
    v2f a, b;
    a.x = A[(size_t)mrow * K + kk + kb];
    a.y = A[(size_t)mrow * K + kk + kb + 1];
    b.x = Bm[(size_t)(kk + kb)     * Nc + col];
    b.y = Bm[(size_t)(kk + kb + 1) * Nc + col];
    acc = __builtin_amdgcn_wmma_f32_16x16x4_f32(false, a, false, b, (short)0, acc, false, false);
  }
  const int r0 = (lane >> 4) * 8;
#pragma unroll
  for (int r = 0; r < 8; ++r)
    C[(size_t)(tm * 16 + r0 + r) * Nc + col] = acc[r];
}

// ---------------------------------------------------------------------------
// s_i / s_j score dots; one block per node row, one wave per head.
// ---------------------------------------------------------------------------
__global__ void attn_scores(const float* __restrict__ hbuf, const float* __restrict__ attn,
                            float* __restrict__ si, float* __restrict__ sj, int Fh){
  const int row  = blockIdx.x;
  const int head = threadIdx.x >> 5;
  const int lane = threadIdx.x & 31;
  const int AP = 2 * Fh + 1;
  float vi = 0.f, vj = 0.f;
  for (int f = lane; f < Fh; f += 32){
    float hv = hbuf[((size_t)row * HH + head) * Fh + f];
    vi += hv * attn[head * AP + f];
    vj += hv * attn[head * AP + Fh + f];
  }
#pragma unroll
  for (int m = 16; m; m >>= 1){
    vi += __shfl_xor(vi, m, 32);
    vj += __shfl_xor(vj, m, 32);
  }
  if (lane == 0){
    si[(size_t)row * HH + head] = vi;
    sj[(size_t)row * HH + head] = vj;
  }
}

// ---------------------------------------------------------------------------
// Transpose + cast: Hbuf[b,n,h,f] (f32) -> hbfT[b,h,f,n] (bf16).
// ---------------------------------------------------------------------------
__global__ void transpose_bf16(const float* __restrict__ in, unsigned short* __restrict__ out,
                               int Fh){
  const int tx = threadIdx.x, ty = threadIdx.y;
  const int bh = blockIdx.z;
  const int b = bh >> 2, head = bh & 3;
  const int n0 = blockIdx.x * 32, f0 = blockIdx.y * 32;
  __shared__ unsigned short tile[32][33];
#pragma unroll
  for (int r = 0; r < 4; ++r){
    int nl = ty + r * 8;
    tile[nl][tx] = f2bf(in[((size_t)(b * NN + n0 + nl) * HH + head) * Fh + f0 + tx]);
  }
  __syncthreads();
#pragma unroll
  for (int r = 0; r < 4; ++r){
    int fl = ty + r * 8;
    out[((size_t)(b * HH + head) * Fh + f0 + fl) * NN + n0 + tx] = tile[tx][fl];
  }
}

// ---------------------------------------------------------------------------
// Streaming softmax stats per (b,i,h): row max m, denom l. Never builds e.
// ---------------------------------------------------------------------------
__global__ void softmax_stats(const float* __restrict__ adj, const float* __restrict__ si,
                              const float* __restrict__ sj, const float* __restrict__ attn,
                              float* __restrict__ Mv, float* __restrict__ Lv, int Fh){
  const int b = blockIdx.y, i = blockIdx.x, t = threadIdx.x;
  const int AP = 2 * Fh + 1;
  float ae[HH], sih[HH];
#pragma unroll
  for (int h = 0; h < HH; ++h){
    ae[h]  = attn[h * AP + 2 * Fh];
    sih[h] = si[((size_t)b * NN + i) * HH + h];
  }
  float m[HH], l[HH];
#pragma unroll
  for (int h = 0; h < HH; ++h){ m[h] = -__builtin_inff(); l[h] = 0.f; }
  const float* arow = adj + ((size_t)b * NN + i) * NN;
  for (int j = t; j < NN; j += 256){
    if (j + 256 < NN) __builtin_prefetch(arow + j + 256, 0, 1);
    float av = arow[j];
    if (av != 0.f){
      const float* sjr = sj + ((size_t)b * NN + j) * HH;
#pragma unroll
      for (int h = 0; h < HH; ++h){
        float e = sih[h] + sjr[h] + av * ae[h];
        e = e < 0.f ? NSLOPE * e : e;
        if (e > m[h]){ l[h] = l[h] * __expf(m[h] - e) + 1.f; m[h] = e; }
        else          { l[h] += __expf(e - m[h]); }
      }
    }
  }
  __shared__ float sm[HH][256], sl[HH][256];
#pragma unroll
  for (int h = 0; h < HH; ++h){ sm[h][t] = m[h]; sl[h][t] = l[h]; }
  __syncthreads();
  for (int off = 128; off; off >>= 1){
    if (t < off){
#pragma unroll
      for (int h = 0; h < HH; ++h){
        float m1 = sm[h][t], m2 = sm[h][t + off];
        float l1 = sl[h][t], l2 = sl[h][t + off];
        float mm = fmaxf(m1, m2), ll = 0.f;
        if (l1 > 0.f) ll += l1 * __expf(m1 - mm);
        if (l2 > 0.f) ll += l2 * __expf(m2 - mm);
        sm[h][t] = mm; sl[h][t] = ll;
      }
    }
    __syncthreads();
  }
  if (t == 0){
#pragma unroll
    for (int h = 0; h < HH; ++h){
      Mv[((size_t)b * NN + i) * HH + h] = sm[h][0];
      Lv[((size_t)b * NN + i) * HH + h] = sl[h][0];
    }
  }
}

// ---------------------------------------------------------------------------
// Fused attention aggregation: out[b,i,h,:] = sum_j alpha * h[j,:]
// 4 waves/block (one head each), 16-row i tile, j streamed in chunks of 32.
// Double-buffered TDM: chunk jc+1 DMAs into the alternate LDS buffer while
// chunk jc's alpha tile + 16x16x32 bf16 WMMAs execute (wait tensorcnt<=1).
// WA: alpha tile staged in LDS and written out as contiguous b128 stores.
// ---------------------------------------------------------------------------
template<int FH, bool WA>
__global__ void attn_aggregate(const unsigned short* __restrict__ hbfT, // [b,h,f,n] bf16
                               const float* __restrict__ adj,
                               const float* __restrict__ si,
                               const float* __restrict__ sj,
                               const float* __restrict__ attn,
                               const float* __restrict__ Mv,
                               const float* __restrict__ Lv,
                               float* __restrict__ outAgg,
                               float* __restrict__ outAlpha){
  const int b = blockIdx.y, it = blockIdx.x;
  const int head = threadIdx.x >> 5, lane = threadIdx.x & 31;
  const int tid = threadIdx.x;
  const int AP = 2 * FH + 1;
  const float ae = attn[head * AP + 2 * FH];
  __shared__ __align__(16) unsigned short hBT[2][HH][FH][32]; // B tiles: [f][j] bf16
  __shared__ __align__(16) unsigned short aT[HH][16][32];     // alpha tiles: [i][j] bf16
  constexpr int ASZ = WA ? 16 : 1;
  __shared__ __align__(16) float alphaS[ASZ][32][HH];         // f32 alpha staging (WA)

  float mi[16], il[16], siv[16];
#pragma unroll
  for (int r = 0; r < 16; ++r){
    size_t idx = ((size_t)b * NN + it * 16 + r) * HH + head;
    mi[r]  = Mv[idx];
    float lv = Lv[idx];
    il[r]  = lv > 0.f ? 1.f / lv : 0.f;   // nan_to_num: fully-masked row -> alpha 0
    siv[r] = si[idx];
  }

  constexpr int NT = FH / 16;
  constexpr int NCH = NN / 32;
  const v8f vzero = {0.f,0.f,0.f,0.f,0.f,0.f,0.f,0.f};
  v8f acc[NT];
#pragma unroll
  for (int ft = 0; ft < NT; ++ft) acc[ft] = vzero;

  const int M0 = lane & 15, kb = (lane >> 4) * 16;   // A frag lane mapping
  const int Ncol = lane & 15, ko = (lane >> 4) * 16; // B frag lane mapping

  const unsigned short* hbase = hbfT + (size_t)(b * HH + head) * FH * NN;
  const float* adjBase = adj + ((size_t)b * NN + it * 16) * NN;

#if defined(HAVE_TDM)
  const unsigned ldsOff0 = (unsigned)(unsigned long long)(uintptr_t)&hBT[0][head][0][0];
  const unsigned ldsOff1 = (unsigned)(unsigned long long)(uintptr_t)&hBT[1][head][0][0];
  tdm_issue<FH>(hbase, ldsOff0);                    // prologue: chunk 0 -> buf 0
#endif

  for (int jc = 0; jc < NCH; ++jc){
    const int cur = jc & 1;
#if defined(HAVE_TDM)
    if (jc + 1 < NCH){ // DMA next chunk into the other buffer
      asm volatile("s_wait_dscnt 0x0" ::: "memory"); // prior reads of that buffer done
      tdm_issue<FH>(hbase + (jc + 1) * 32, cur ? ldsOff0 : ldsOff1);
    }
#else
    { // fallback: manual staging of chunk jc
#pragma unroll
      for (int q = 0; q < FH / 8; ++q){
        int idx = q * 32 + lane;
        int f = idx >> 2, c = idx & 3;
        ((uint4*)&hBT[cur][head][0][0])[idx] =
            *(const uint4*)(hbase + (size_t)f * NN + jc * 32 + c * 8);
      }
    }
#endif
    const int j = jc * 32 + lane;
    const float sjv = sj[((size_t)b * NN + j) * HH + head];
    if (jc + 1 < NCH){ // prefetch next adj chunk (4 rows per wave)
#pragma unroll
      for (int p = 0; p < 4; ++p)
        __builtin_prefetch(adjBase + (size_t)(head * 4 + p) * NN + (jc + 1) * 32 + lane, 0, 1);
    }
    // alpha tile (overlaps the tensor DMA)
#pragma unroll
    for (int r = 0; r < 16; ++r){
      float av = adjBase[(size_t)r * NN + j];
      float e = siv[r] + sjv + av * ae;
      e = e < 0.f ? NSLOPE * e : e;
      float al = (av == 0.f) ? 0.f : __expf(e - mi[r]) * il[r];
      aT[head][r][lane] = f2bf(al);
      if (WA) alphaS[WA ? r : 0][lane][head] = al;
    }
    if (WA){ // cooperative contiguous alpha write-out (512B per i row)
      __syncthreads();
      float* abase = outAlpha + (((size_t)b * NN + it * 16) * NN + jc * 32) * HH;
      const float4* aS = (const float4*)&alphaS[0][0][0];
#pragma unroll
      for (int q4 = 0; q4 < 4; ++q4){
        int q = q4 * 128 + tid;          // 512 float4 chunks total
        int i = q >> 5, c = q & 31;      // 32 float4 per i row
        ((float4*)(abase + (size_t)i * NN * HH))[c] = aS[q];
      }
      __syncthreads();
    }
    // A fragment: lane<16 -> K 0..15 ; lane>=16 -> K 16..31
    BF16x16 af;
    af.q[0] = *(const uint4*)&aT[head][M0][kb];
    af.q[1] = *(const uint4*)&aT[head][M0][kb + 8];
#if defined(HAVE_TDM)
    if (jc + 1 < NCH) __builtin_amdgcn_s_wait_tensorcnt((unsigned short)1); // buf cur ready
    else              __builtin_amdgcn_s_wait_tensorcnt((unsigned short)0);
#endif
#pragma unroll
    for (int ft = 0; ft < NT; ++ft){
      BF16x16 bm; // B fragment: lanes0-15 K0..15, lanes16-31 K16..31, N = f
      const uint4* bp = (const uint4*)&hBT[cur][head][ft * 16 + Ncol][ko];
      bm.q[0] = bp[0];
      bm.q[1] = bp[1];
      acc[ft] = __builtin_amdgcn_wmma_f32_16x16x32_bf16(false, af.v, false, bm.v,
                                                        (short)0, acc[ft], false, false);
    }
  }
  const int r0 = (lane >> 4) * 8;
#pragma unroll
  for (int ft = 0; ft < NT; ++ft)
#pragma unroll
    for (int r = 0; r < 8; ++r)
      outAgg[(((size_t)b * NN + it * 16 + r0 + r) * HH + head) * FH + ft * 16 + Ncol] = acc[ft][r];
}

// ---------------------------------------------------------------------------
// Epilogue: optional head-mean, + residual, LayerNorm(128), optional ELU.
// ---------------------------------------------------------------------------
__global__ void epilogue_ln(const float* __restrict__ agg, const float* __restrict__ res,
                            const float* __restrict__ gam, const float* __restrict__ bet,
                            float* __restrict__ out, int meanHeads, int doElu){
  const int row = blockIdx.x, t = threadIdx.x; // F = 128
  float v;
  if (meanHeads){
    v = 0.25f * (agg[((size_t)row * HH + 0) * 128 + t] + agg[((size_t)row * HH + 1) * 128 + t]
               + agg[((size_t)row * HH + 2) * 128 + t] + agg[((size_t)row * HH + 3) * 128 + t]);
  } else {
    v = agg[(size_t)row * 128 + t];
  }
  v += res[(size_t)row * 128 + t];
  __shared__ float red[128];
  red[t] = v; __syncthreads();
  for (int s = 64; s; s >>= 1){ if (t < s) red[t] += red[t + s]; __syncthreads(); }
  float mean = red[0] * (1.f / 128.f); __syncthreads();
  float d = v - mean;
  red[t] = d * d; __syncthreads();
  for (int s = 64; s; s >>= 1){ if (t < s) red[t] += red[t + s]; __syncthreads(); }
  float var = red[0] * (1.f / 128.f);
  float y = d * rsqrtf(var + EPSLN) * gam[t] + bet[t];
  if (doElu) y = y > 0.f ? y : __expf(y) - 1.f;
  out[(size_t)row * 128 + t] = y;
}

// ---------------------------------------------------------------------------
// Attention pooling: scores = Z@pw+pb, softmax over nodes, g = sum w*Z.
// ---------------------------------------------------------------------------
__global__ void pool_kernel(const float* __restrict__ Z, const float* __restrict__ pw,
                            const float* __restrict__ pb, float* __restrict__ g){
  const int b = blockIdx.x, t = threadIdx.x;
  __shared__ float sc[NN];
  __shared__ float red[128];
  for (int n = t; n < NN; n += 128){
    float dot = pb[0];
    for (int f = 0; f < 128; ++f) dot += Z[((size_t)b * NN + n) * 128 + f] * pw[f];
    sc[n] = dot;
  }
  __syncthreads();
  float mx = -__builtin_inff();
  for (int n = t; n < NN; n += 128) mx = fmaxf(mx, sc[n]);
  red[t] = mx; __syncthreads();
  for (int s = 64; s; s >>= 1){ if (t < s) red[t] = fmaxf(red[t], red[t + s]); __syncthreads(); }
  mx = red[0]; __syncthreads();
  float sum = 0.f;
  for (int n = t; n < NN; n += 128) sum += __expf(sc[n] - mx);
  red[t] = sum; __syncthreads();
  for (int s = 64; s; s >>= 1){ if (t < s) red[t] += red[t + s]; __syncthreads(); }
  sum = red[0];
  float acc = 0.f;
  for (int n = 0; n < NN; ++n) acc += __expf(sc[n] - mx) * Z[((size_t)b * NN + n) * 128 + t];
  g[(size_t)b * 128 + t] = acc / sum;
}

// ---------------------------------------------------------------------------
extern "C" void kernel_launch(void* const* d_in, const int* in_sizes, int n_in,
                              void* d_out, int out_size, void* d_ws, size_t ws_size,
                              hipStream_t stream){
  const float* x      = (const float*)d_in[0];
  const float* adj    = (const float*)d_in[1];
  const float* W1     = (const float*)d_in[2];
  const float* a1     = (const float*)d_in[3];
  const float* res1_w = (const float*)d_in[4];
  const float* ln1_g  = (const float*)d_in[5];
  const float* ln1_b  = (const float*)d_in[6];
  const float* W2     = (const float*)d_in[7];
  const float* a2     = (const float*)d_in[8];
  const float* res2_w = (const float*)d_in[9];
  const float* ln2_g  = (const float*)d_in[10];
  const float* ln2_b  = (const float*)d_in[11];
  const float* W3     = (const float*)d_in[12];
  const float* a3     = (const float*)d_in[13];
  const float* ln3_g  = (const float*)d_in[14];
  const float* ln3_b  = (const float*)d_in[15];
  const float* pw     = (const float*)d_in[16];
  const float* pb     = (const float*)d_in[17];

  char* ws = (char*)d_ws;
  const size_t MB = 1024u * 1024u;
  const size_t R  = (size_t)BB * NN;                    // 8192 rows
  float*          Hbuf = (float*)(ws);                  // [R,512] fp32 (16 MB max)
  unsigned short* HbfT = (unsigned short*)(ws + 16*MB); // [b,h,f,n] bf16 (8 MB max)
  float*          Res  = (float*)(ws + 24*MB);          // [R,128]
  float*          X1   = (float*)(ws + 28*MB);          // [R,128]
  float*          X2   = (float*)(ws + 32*MB);          // [R,128]
  float*          Agg  = (float*)(ws + 36*MB);          // [R,4,128] (16 MB max)
  float*          Si   = (float*)(ws + 52*MB);
  float*          Sj   = (float*)(ws + 52*MB + 128*1024);
  float*          Mv   = (float*)(ws + 52*MB + 256*1024);
  float*          Lv   = (float*)(ws + 52*MB + 384*1024);

  float* Zout = (float*)d_out;          // [8,1024,128]
  float* Gout = Zout + R * 128;         // [8,128]
  float* Aout = Gout + (size_t)BB * 128;// [8,1024,1024,4]

  dim3 blk128(128), blk256(256), blkT(32, 8);
  const unsigned TM = (unsigned)(R / 16);

  // ---- Layer 1 (Fh = 32, concat heads) ----
  gemm_f32_wmma <<<dim3(TM, 2), blk128, 0, stream>>>(x, W1, Hbuf, DD, 128);
  gemm_f32_wmma <<<dim3(TM, 2), blk128, 0, stream>>>(x, res1_w, Res, DD, 128);
  attn_scores   <<<dim3((unsigned)R), blk128, 0, stream>>>(Hbuf, a1, Si, Sj, 32);
  transpose_bf16<<<dim3(NN/32, 1, BB*HH), blkT, 0, stream>>>(Hbuf, HbfT, 32);
  softmax_stats <<<dim3(NN, BB), blk256, 0, stream>>>(adj, Si, Sj, a1, Mv, Lv, 32);
  attn_aggregate<32,false><<<dim3(NN/16, BB), blk128, 0, stream>>>(HbfT, adj, Si, Sj, a1, Mv, Lv, Agg, nullptr);
  epilogue_ln   <<<dim3((unsigned)R), blk128, 0, stream>>>(Agg, Res, ln1_g, ln1_b, X1, 0, 1);

  // ---- Layer 2 (Fh = 128, head mean) ----
  gemm_f32_wmma <<<dim3(TM, 8), blk128, 0, stream>>>(X1, W2, Hbuf, 128, 512);
  gemm_f32_wmma <<<dim3(TM, 2), blk128, 0, stream>>>(X1, res2_w, Res, 128, 128);
  attn_scores   <<<dim3((unsigned)R), blk128, 0, stream>>>(Hbuf, a2, Si, Sj, 128);
  transpose_bf16<<<dim3(NN/32, 4, BB*HH), blkT, 0, stream>>>(Hbuf, HbfT, 128);
  softmax_stats <<<dim3(NN, BB), blk256, 0, stream>>>(adj, Si, Sj, a2, Mv, Lv, 128);
  attn_aggregate<128,false><<<dim3(NN/16, BB), blk128, 0, stream>>>(HbfT, adj, Si, Sj, a2, Mv, Lv, Agg, nullptr);
  epilogue_ln   <<<dim3((unsigned)R), blk128, 0, stream>>>(Agg, Res, ln2_g, ln2_b, X2, 1, 0);

  // ---- Layer 3 (Fh = 128, head mean, alpha is an output) ----
  gemm_f32_wmma <<<dim3(TM, 8), blk128, 0, stream>>>(X2, W3, Hbuf, 128, 512);
  attn_scores   <<<dim3((unsigned)R), blk128, 0, stream>>>(Hbuf, a3, Si, Sj, 128);
  transpose_bf16<<<dim3(NN/32, 4, BB*HH), blkT, 0, stream>>>(Hbuf, HbfT, 128);
  softmax_stats <<<dim3(NN, BB), blk256, 0, stream>>>(adj, Si, Sj, a3, Mv, Lv, 128);
  attn_aggregate<128,true><<<dim3(NN/16, BB), blk128, 0, stream>>>(HbfT, adj, Si, Sj, a3, Mv, Lv, Agg, Aout);
  epilogue_ln   <<<dim3((unsigned)R), blk128, 0, stream>>>(Agg, X2, ln3_g, ln3_b, Zout, 1, 0);

  // ---- Pool ----
  pool_kernel   <<<dim3(BB), blk128, 0, stream>>>(Zout, pw, pb, Gout);
}
